// MinGRUCell_51273319579651
// MI455X (gfx1250) — compile-verified, matrix operand order checked by reading, not run on hardware
//
#include <hip/hip_runtime.h>

// ---- CDNA5 WMMA types -------------------------------------------------------
typedef __attribute__((ext_vector_type(16))) __bf16 v16bf;
typedef __attribute__((ext_vector_type(8)))  __bf16 v8bf;
typedef __attribute__((ext_vector_type(4)))  __bf16 v4bf;
typedef __attribute__((ext_vector_type(8)))  float  v8f;

#define LDS_STRIDE 40   // 32 K-elements + 8 bf16 pad (80B row) -> conflict-free b128 reads

union Frag16 { struct { v8bf lo, hi; } p; v16bf v; };

// Load a 16x32 bf16 fragment (A-layout, which also serves as the B operand for
// C = X * W^T since rows of W are K-major just like rows of X).
// Per ISA 7.12.2: lane<16 holds row m, K=0..7 (v0..3) and K=16..23 (v4..7);
// lane>=16 holds row m, K=8..15 and K=24..31 -> two contiguous 16B LDS reads.
__device__ __forceinline__ v16bf load_frag(const __bf16* __restrict__ tile_row0,
                                           int m, int half) {
    const __bf16* p = tile_row0 + m * LDS_STRIDE;
    Frag16 f;
    f.p.lo = *(const v8bf*)(p + half * 8);        // K =  0..7  or  8..15
    f.p.hi = *(const v8bf*)(p + 16 + half * 8);   // K = 16..23 or 24..31
    return f.v;
}

__device__ __forceinline__ v8f zero8() {
    v8f z;
#pragma unroll
    for (int i = 0; i < 8; ++i) z[i] = 0.0f;
    return z;
}

// ---- Kernel 1: dual-gate GEMM (bf16 WMMA, f32 accumulate) -------------------
// grid = (H/128, BT/64), block = 256 threads (8 waves, 2x4)
// wave tile: 32(M) x 32(N) per gate = 2x2 v_wmma_f32_16x16x32_bf16 tiles.
__global__ __launch_bounds__(256)
void mingru_gemm_kernel(const float* __restrict__ x,
                        const float* __restrict__ Wz,
                        const float* __restrict__ bz,
                        const float* __restrict__ Wh,
                        const float* __restrict__ bh,
                        float* __restrict__ z_out,   // sigmoid gate -> workspace
                        float* __restrict__ ht_out,  // candidate    -> d_out
                        int D, int H)
{
    __shared__ __align__(16) __bf16 lA [ 64 * LDS_STRIDE];
    __shared__ __align__(16) __bf16 lBz[128 * LDS_STRIDE];
    __shared__ __align__(16) __bf16 lBh[128 * LDS_STRIDE];

    const int tid   = threadIdx.x;
    const int lane  = tid & 31;
    const int wid   = tid >> 5;
    const int waveM = wid >> 2;     // 0..1
    const int waveN = wid & 3;      // 0..3
    const int m     = lane & 15;
    const int half  = lane >> 4;

    const int tileM = blockIdx.y * 64;
    const int tileN = blockIdx.x * 128;

    v8f accZ[2][2], accH[2][2];
#pragma unroll
    for (int a = 0; a < 2; ++a)
#pragma unroll
        for (int b = 0; b < 2; ++b) { accZ[a][b] = zero8(); accH[a][b] = zero8(); }

    // staging decomposition: 8 float4 column-groups per 32-row slab
    const int sr = tid >> 3;          // 0..31 row within slab
    const int sc = (tid & 7) * 4;     // K column (float4 granularity)

    for (int kk = 0; kk < D; kk += 32) {
        // ---- stage fp32 -> bf16 tiles into LDS ----
#pragma unroll
        for (int rr = 0; rr < 64; rr += 32) {
            const float4 f = *(const float4*)&x[(size_t)(tileM + sr + rr) * D + kk + sc];
            v4bf c; c[0] = (__bf16)f.x; c[1] = (__bf16)f.y;
                    c[2] = (__bf16)f.z; c[3] = (__bf16)f.w;
            *(v4bf*)&lA[(sr + rr) * LDS_STRIDE + sc] = c;
        }
#pragma unroll
        for (int rr = 0; rr < 128; rr += 32) {
            const size_t wrow = (size_t)(tileN + sr + rr) * D + kk + sc;
            float4 fz = *(const float4*)&Wz[wrow];
            v4bf cz; cz[0] = (__bf16)fz.x; cz[1] = (__bf16)fz.y;
                     cz[2] = (__bf16)fz.z; cz[3] = (__bf16)fz.w;
            *(v4bf*)&lBz[(sr + rr) * LDS_STRIDE + sc] = cz;
            float4 fh = *(const float4*)&Wh[wrow];
            v4bf ch; ch[0] = (__bf16)fh.x; ch[1] = (__bf16)fh.y;
                     ch[2] = (__bf16)fh.z; ch[3] = (__bf16)fh.w;
            *(v4bf*)&lBh[(sr + rr) * LDS_STRIDE + sc] = ch;
        }
        __syncthreads();

        // ---- fragments + 8 WMMAs (2 gates x 2x2 tiles) ----
        v16bf aF[2];
        aF[0] = load_frag(&lA[(waveM * 32 +  0) * LDS_STRIDE], m, half);
        aF[1] = load_frag(&lA[(waveM * 32 + 16) * LDS_STRIDE], m, half);
#pragma unroll
        for (int tn = 0; tn < 2; ++tn) {
            const v16bf bZ = load_frag(&lBz[(waveN * 32 + tn * 16) * LDS_STRIDE], m, half);
            const v16bf bH = load_frag(&lBh[(waveN * 32 + tn * 16) * LDS_STRIDE], m, half);
#pragma unroll
            for (int tm = 0; tm < 2; ++tm) {
                accZ[tm][tn] = __builtin_amdgcn_wmma_f32_16x16x32_bf16(
                    false, aF[tm], false, bZ, (short)0, accZ[tm][tn], false, false);
                accH[tm][tn] = __builtin_amdgcn_wmma_f32_16x16x32_bf16(
                    false, aF[tm], false, bH, (short)0, accH[tm][tn], false, false);
            }
        }
        __syncthreads();
    }

    // ---- epilogue: bias, sigmoid(z), scattered-but-half-wave-coalesced stores
    // C layout (ISA 7.12.2): VGPR j -> row (j + half*8), col = lane&15.
#pragma unroll
    for (int tn = 0; tn < 2; ++tn) {
        const int gn  = tileN + waveN * 32 + tn * 16 + m;
        const float bzv = bz[gn];
        const float bhv = bh[gn];
#pragma unroll
        for (int tm = 0; tm < 2; ++tm) {
            const int gm0 = tileM + waveM * 32 + tm * 16 + half * 8;
#pragma unroll
            for (int j = 0; j < 8; ++j) {
                const size_t idx = (size_t)(gm0 + j) * H + gn;
                float zv = accZ[tm][tn][j] + bzv;
                zv = 1.0f / (1.0f + __expf(-zv));
                z_out[idx]  = zv;
                ht_out[idx] = accH[tm][tn][j] + bhv;
            }
        }
    }
}

// ---- Kernel 2: sequential affine scan over T, parallel over (b,h) -----------
// out holds h_tilde on entry and is overwritten in place with h_t.
__global__ __launch_bounds__(256)
void mingru_scan_kernel(const float* __restrict__ z,
                        const float* __restrict__ h0,
                        float* __restrict__ out,
                        int B, int T, int H)
{
    const int idx = blockIdx.x * blockDim.x + threadIdx.x;
    if (idx >= B * H) return;
    const int b = idx / H;
    const int h = idx - b * H;

    float hv = h0[idx];                       // h_prev is (B,H) flat
    size_t p = (size_t)b * T * H + h;
    for (int t = 0; t < T; ++t, p += H) {
        const float zt = z[p];
        const float ht = out[p];
        hv = (1.0f - zt) * hv + zt * ht;
        out[p] = hv;
    }
}

// ---- Launch -----------------------------------------------------------------
extern "C" void kernel_launch(void* const* d_in, const int* in_sizes, int n_in,
                              void* d_out, int out_size, void* d_ws, size_t ws_size,
                              hipStream_t stream) {
    const float* x   = (const float*)d_in[0];
    const float* h0  = (const float*)d_in[1];
    const float* Wz  = (const float*)d_in[2];
    const float* bz  = (const float*)d_in[3];
    const float* Wh  = (const float*)d_in[4];
    const float* bh  = (const float*)d_in[5];
    float* out = (float*)d_out;

    const int H  = in_sizes[3];          // 1024
    const int D  = in_sizes[2] / H;      // 1024
    const int BT = in_sizes[0] / D;      // 32768
    const int B  = in_sizes[1] / H;      // 16
    const int T  = BT / B;               // 2048

    float* z_ws = (float*)d_ws;          // B*T*H fp32 gate buffer (fully overwritten)

    dim3 grid(H / 128, BT / 64);
    mingru_gemm_kernel<<<grid, 256, 0, stream>>>(x, Wz, bz, Wh, bh, z_ws, out, D, H);

    const int nscan = B * H;
    mingru_scan_kernel<<<(nscan + 255) / 256, 256, 0, stream>>>(z_ws, h0, out, B, T, H);
}
